// MllamaTextSelfAttention_738734375673
// MI455X (gfx1250) — matrix operand order repaired and checked by reading, not compiled, not run on hardware
//
#include <hip/hip_runtime.h>
#include <hip/hip_bf16.h>

typedef __attribute__((ext_vector_type(16))) __bf16       v16bf;
typedef __attribute__((ext_vector_type(8)))  float        v8f;
typedef __attribute__((ext_vector_type(4)))  unsigned int v4u;
typedef __attribute__((ext_vector_type(8)))  int          v8i;
typedef __attribute__((ext_vector_type(4)))  int          v4i;

#define S_LEN   2048
#define NHEADS  32
#define KVHEADS 8
#define HDIM    128
#define MODEL   (NHEADS * HDIM)    // 4096
#define KVMODEL (KVHEADS * HDIM)   // 1024

#define HAVE_TDM   __has_builtin(__builtin_amdgcn_tensor_load_to_lds)
#define HAVE_ASYNC (__has_builtin(__builtin_amdgcn_global_load_async_to_lds_b128) && \
                    __has_builtin(__builtin_amdgcn_s_wait_asynccnt))
#if HAVE_TDM
#define USE_TDM 1
#define USE_ASYNC 0
#elif HAVE_ASYNC
#define USE_TDM 0
#define USE_ASYNC 1
#else
#define USE_TDM 0
#define USE_ASYNC 0
#endif

// ---- compile-log probes ----
#define XSTR(x) #x
#define SSTR(x) XSTR(x)
#pragma message("CDNA5-PROBE clang_major=" SSTR(__clang_major__))
#if USE_TDM
#pragma message("CDNA5-PROBE staging path: TDM tensor_load_to_lds")
#elif USE_ASYNC
#pragma message("CDNA5-PROBE staging path: global_load_async_to_lds_b128")
#else
#pragma message("CDNA5-PROBE staging path: plain LDS copies")
#endif

__device__ inline __bf16 to_bf16(float x)  { return (__bf16)x; }
__device__ inline __bf16 to_bf16(__bf16 x) { return x; }

__device__ inline v8f wmma_bf16(const v16bf& a, const v16bf& b, const v8f& c) {
  // D = A(16x32 bf16) x B(32x16 bf16) + C(16x16 f32)
  return __builtin_amdgcn_wmma_f32_16x16x32_bf16(
      false, a, false, b, (short)0, c, false, false);
}

__device__ inline v8f v8f_zero() {
  v8f z = {0.f, 0.f, 0.f, 0.f, 0.f, 0.f, 0.f, 0.f};
  return z;
}

#if USE_TDM
__device__ __forceinline__ void tdm_issue(v4u g0, v8i g1) {
  v4i z4 = {0, 0, 0, 0};
#if defined(__clang_major__) && __clang_major__ >= 23
  v8i z8 = {0, 0, 0, 0, 0, 0, 0, 0};
  __builtin_amdgcn_tensor_load_to_lds(g0, g1, z4, z4, z8, 0);
#else
  __builtin_amdgcn_tensor_load_to_lds(g0, g1, z4, z4, 0);
#endif
}

// 2D tile DMA: tile_d0 contiguous elements per row, tile_d1 rows,
// row stride stride0 (elements), element size = 1<<ds_log2 bytes.
// Packs the Tensor DMA Descriptor per CDNA5 ISA section 8.3/8.4.
__device__ __forceinline__ void tdm_load_2d(const void* gptr, void* ldsptr,
                                            int ds_log2, int tile_d0, int tile_d1,
                                            long long stride0)
{
  unsigned long long ga = (unsigned long long)(size_t)gptr;
  unsigned lds = (unsigned)(size_t)ldsptr;   // low 32 bits of flat LDS addr = LDS offset
  v4u g0;
  g0[0] = 1u;                                            // count=1, user descriptor
  g0[1] = lds;                                           // lds_addr
  g0[2] = (unsigned)ga;                                  // global_addr[31:0]
  g0[3] = (unsigned)((ga >> 32) & 0x01ffffffu) | (2u << 30); // global_addr[56:32] | type=2
  v8i g1;
  g1[0] = (ds_log2 & 3) << 16;                           // workgroup_mask=0 | data_size
  g1[1] = (tile_d0 & 0xffff) << 16;                      // tensor_dim0[15:0]
  g1[2] = ((unsigned)tile_d0 >> 16) | ((tile_d1 & 0xffff) << 16); // td0 hi | tensor_dim1 lo
  g1[3] = ((unsigned)tile_d1 >> 16) | ((tile_d0 & 0xffff) << 16); // td1 hi | tile_dim0
  g1[4] = (tile_d1 & 0xffff);                            // tile_dim1 | tile_dim2=0
  g1[5] = (int)(stride0 & 0xffffffffll);                 // tensor_dim0_stride[31:0]
  g1[6] = (int)((stride0 >> 32) & 0xffffll);             // stride[47:32] | dim1_stride=0
  g1[7] = 0;
  tdm_issue(g0, g1);
}
#endif

#if USE_ASYNC
// 16-byte async global->LDS chunk (b128 => <4 x i32> typed pointers).
__device__ __forceinline__ void async_cp16(const void* g, void* l) {
  __builtin_amdgcn_global_load_async_to_lds_b128((v4i*)g, (v4i*)l, 0, 0);
}
#endif

// -----------------------------------------------------------------------------
// Tiled GEMM: C[m,n] = sum_k A[m,k] * B[n,k]
// Block tile 128x128, 8 waves (wave32), each wave computes 32x64.
// A/B 128x32 K-slabs staged in LDS, double buffered, DMA-loaded.
// -----------------------------------------------------------------------------
template<typename AT>
__global__ void __launch_bounds__(256)
gemm_bt_tiled_kernel(const AT* __restrict__ A, const float* __restrict__ B,
                     float* __restrict__ C, int M, int N, int K)
{
  __shared__ AT    As[2][128][32];
  __shared__ float Bs[2][128][32];

  const int tid  = threadIdx.x;
  const int lane = tid & 31;
  const int wave = tid >> 5;
  const int l16  = lane & 15;
  const int half = lane >> 4;
  const int mBase = blockIdx.y * 128;
  const int nBase = blockIdx.x * 128;
  const int mOff  = (wave >> 1) * 32;   // 4 wave-rows x 32
  const int nOff  = (wave & 1) * 64;    // 2 wave-cols x 64

  v8f acc[2][4];
  #pragma unroll
  for (int i = 0; i < 2; ++i)
    #pragma unroll
    for (int t = 0; t < 4; ++t) acc[i][t] = v8f_zero();

  auto issue = [&](int b, int kb) {
#if USE_TDM
    if (wave == 0) {
      tdm_load_2d(A + (size_t)mBase * K + kb, (void*)&As[b][0][0],
                  (sizeof(AT) == 2) ? 1 : 2, 32, 128, K);
      tdm_load_2d(B + (size_t)nBase * K + kb, (void*)&Bs[b][0][0],
                  2, 32, 128, K);
    }
#elif USE_ASYNC
    {
      const int cprA = (int)(32 * sizeof(AT)) / 16;       // 16B chunks per A row
      for (int i = tid; i < 128 * cprA; i += 256) {
        const int r = i / cprA, cb = (i % cprA) * 16;
        async_cp16((const char*)(A + (size_t)(mBase + r) * K + kb) + cb,
                   (char*)&As[b][r][0] + cb);
      }
      for (int i = tid; i < 128 * 8; i += 256) {
        const int r = i >> 3, cb = (i & 7) * 16;
        async_cp16((const char*)(B + (size_t)(nBase + r) * K + kb) + cb,
                   (char*)&Bs[b][r][0] + cb);
      }
    }
#else
    for (int i = tid; i < 128 * 32; i += 256) {
      const int r = i >> 5, c = i & 31;
      As[b][r][c] = A[(size_t)(mBase + r) * K + kb + c];
      Bs[b][r][c] = B[(size_t)(nBase + r) * K + kb + c];
    }
#endif
  };
  auto wait_publish = [&]() {
#if USE_TDM
    if (wave == 0) __builtin_amdgcn_s_wait_tensorcnt(0);
#elif USE_ASYNC
    __builtin_amdgcn_s_wait_asynccnt(0);
#endif
    __syncthreads();
  };

  issue(0, 0);
  wait_publish();

  int buf = 0;
  for (int kb = 0; kb < K; kb += 32) {
    if (kb + 32 < K) issue(buf ^ 1, kb + 32);

    v16bf am[2], bn[4];
    #pragma unroll
    for (int i = 0; i < 2; ++i) {
      const AT* ar = &As[buf][mOff + i * 16 + l16][0];
      #pragma unroll
      for (int j = 0; j < 8; ++j) {
        am[i][j]     = to_bf16(ar[8 * half + j]);
        am[i][j + 8] = to_bf16(ar[16 + 8 * half + j]);
      }
    }
    #pragma unroll
    for (int t = 0; t < 4; ++t) {
      const float* br = &Bs[buf][nOff + t * 16 + l16][0];
      #pragma unroll
      for (int j = 0; j < 16; ++j) bn[t][j] = to_bf16(br[(j < 8) ? (8 * half + j) : (16 + 8 * half + j - 8)]);
    }
    #pragma unroll
    for (int i = 0; i < 2; ++i)
      #pragma unroll
      for (int t = 0; t < 4; ++t)
        acc[i][t] = wmma_bf16(am[i], bn[t], acc[i][t]);

    wait_publish();      // next slab ready; everyone done reading current slab
    buf ^= 1;
  }

  #pragma unroll
  for (int i = 0; i < 2; ++i)
    #pragma unroll
    for (int t = 0; t < 4; ++t)
      #pragma unroll
      for (int r = 0; r < 8; ++r)
        C[(size_t)(mBase + mOff + i * 16 + r + 8 * half) * N +
          (nBase + nOff + t * 16 + l16)] = acc[i][t][r];
}

// -----------------------------------------------------------------------------
// RoPE (fp32) + bf16 convert. Q gets softmax scale folded in.
// Qb: [h][s][d] bf16 ; Kb: [kvh][s][d] bf16
// -----------------------------------------------------------------------------
__global__ void rope_convert_kernel(const float* __restrict__ qf, const float* __restrict__ kf,
                                    const float* __restrict__ cosb, const float* __restrict__ sinb,
                                    __bf16* __restrict__ Qb, __bf16* __restrict__ Kb,
                                    float qscale)
{
  const int idx = blockIdx.x * blockDim.x + threadIdx.x;
  const int total = S_LEN * (NHEADS + KVHEADS) * (HDIM / 2);
  if (idx >= total) return;
  const int d  = idx & 63;
  const int hh = (idx >> 6) % (NHEADS + KVHEADS);
  const int s  = idx / (64 * (NHEADS + KVHEADS));

  const float c0 = cosb[s * HDIM + d];
  const float c1 = cosb[s * HDIM + d + 64];
  const float s0 = sinb[s * HDIM + d];
  const float s1 = sinb[s * HDIM + d + 64];

  if (hh < NHEADS) {
    const float* src = qf + (size_t)s * MODEL + hh * HDIM;
    const float x0 = src[d], x1 = src[d + 64];
    __bf16* dst = Qb + ((size_t)hh * S_LEN + s) * HDIM;
    dst[d]      = (__bf16)((x0 * c0 - x1 * s0) * qscale);
    dst[d + 64] = (__bf16)((x1 * c1 + x0 * s1) * qscale);
  } else {
    const int kvh = hh - NHEADS;
    const float* src = kf + (size_t)s * KVMODEL + kvh * HDIM;
    const float x0 = src[d], x1 = src[d + 64];
    __bf16* dst = Kb + ((size_t)kvh * S_LEN + s) * HDIM;
    dst[d]      = (__bf16)(x0 * c0 - x1 * s0);
    dst[d + 64] = (__bf16)(x1 * c1 + x0 * s1);
  }
}

// V (f32, [s][kvh*128+d]) -> Vt (bf16, [kvh*128+d][s])
__global__ void v_transpose_kernel(const float* __restrict__ vf, __bf16* __restrict__ Vt)
{
  const int idx = blockIdx.x * blockDim.x + threadIdx.x;
  if (idx >= S_LEN * KVMODEL) return;
  const int s = idx / KVMODEL;
  const int c = idx % KVMODEL;
  Vt[(size_t)c * S_LEN + s] = (__bf16)vf[idx];
}

// -----------------------------------------------------------------------------
// Flash attention. Grid: (S/64, H). Block: 128 threads = 4 waves.
// All 4 waves share one KV head -> K/V chunks DMA-staged in LDS, double buffered.
// Online softmax; P re-laid out C/D -> A fragment through LDS.
// -----------------------------------------------------------------------------
__global__ void __launch_bounds__(128)
flash_attn_kernel(const __bf16* __restrict__ Qb, const __bf16* __restrict__ Kb,
                  const __bf16* __restrict__ Vt, __bf16* __restrict__ Ab)
{
  const int h    = blockIdx.y;
  const int kv   = h >> 2;                 // GQA group of 4
  const int wave = threadIdx.x >> 5;
  const int lane = threadIdx.x & 31;
  const int l16  = lane & 15;
  const int half = lane >> 4;
  const int mTile = blockIdx.x * 64 + wave * 16;

  __shared__ __bf16 Ks[2][32][128];        // 32 keys x 128 d, double buffered
  __shared__ __bf16 Vs[2][128][32];        // 128 d x 32 keys (from Vt)
  __shared__ __bf16 lp[4][16][32];         // per-wave P tile

  auto stage = [&](int b, int kk) {
#if USE_TDM
    if (wave == 0) {
      tdm_load_2d(Kb + ((size_t)kv * S_LEN + kk) * HDIM, (void*)&Ks[b][0][0],
                  1, 128, 32, HDIM);
      tdm_load_2d(Vt + (size_t)kv * HDIM * S_LEN + kk, (void*)&Vs[b][0][0],
                  1, 32, 128, S_LEN);
    }
#elif USE_ASYNC
    for (int i = threadIdx.x; i < 512; i += 128) {
      const int kr = i >> 4, kc = (i & 15) * 8;   // 16 x 16B chunks per 128-bf16 K row
      async_cp16(Kb + ((size_t)kv * S_LEN + kk + kr) * HDIM + kc, &Ks[b][kr][kc]);
      const int vr = i >> 2, vc = (i & 3) * 8;    // 4 x 16B chunks per 32-bf16 V row
      async_cp16(Vt + ((size_t)kv * HDIM + vr) * S_LEN + kk + vc, &Vs[b][vr][vc]);
    }
#else
    for (int i = threadIdx.x; i < 32 * 128; i += 128) {
      const int r = i >> 7, c = i & 127;
      Ks[b][r][c] = Kb[((size_t)kv * S_LEN + kk + r) * HDIM + c];
      const int r2 = i >> 5, c2 = i & 31;
      Vs[b][r2][c2] = Vt[((size_t)kv * HDIM + r2) * S_LEN + kk + c2];
    }
#endif
  };
  auto publish = [&]() {
#if USE_TDM
    if (wave == 0) __builtin_amdgcn_s_wait_tensorcnt(0);
#elif USE_ASYNC
    __builtin_amdgcn_s_wait_asynccnt(0);
#endif
    __syncthreads();
  };

  // Preload Q A-fragments for the full d=128 (4 chunks of K=32)
  const __bf16* qrow = Qb + ((size_t)h * S_LEN + (mTile + l16)) * HDIM;
  v16bf aq[4];
  #pragma unroll
  for (int c = 0; c < 4; ++c) {
    const __bf16* p0 = qrow + 32 * c + 8 * half;
    #pragma unroll
    for (int j = 0; j < 8; ++j) { aq[c][j] = p0[j]; aq[c][j + 8] = p0[16 + j]; }
  }

  v8f oacc[8];
  #pragma unroll
  for (int t = 0; t < 8; ++t) oacc[t] = v8f_zero();
  float mrow[8], lrow[8];
  #pragma unroll
  for (int r = 0; r < 8; ++r) { mrow[r] = -3.0e38f; lrow[r] = 0.f; }

  const int kmax = blockIdx.x * 64 + 64;   // uniform across block (causal)
  stage(0, 0);
  publish();
  int buf = 0;
  for (int kk = 0; kk < kmax; kk += 32) {
    if (kk + 32 < kmax) stage(buf ^ 1, kk + 32);

    // ---- scores: load all 8 K fragments, then 8 WMMAs ----
    v16bf bk0[4], bk1[4];
    #pragma unroll
    for (int c = 0; c < 4; ++c) {
      const __bf16* p0 = &Ks[buf][l16][32 * c + 8 * half];
      const __bf16* p1 = &Ks[buf][16 + l16][32 * c + 8 * half];
      #pragma unroll
      for (int j = 0; j < 8; ++j) { bk0[c][j] = p0[j]; bk0[c][j + 8] = p0[16 + j]; bk1[c][j] = p1[j]; bk1[c][j + 8] = p1[16 + j]; }
    }
    v8f s0 = v8f_zero(), s1 = v8f_zero();
    #pragma unroll
    for (int c = 0; c < 4; ++c) {
      s0 = wmma_bf16(aq[c], bk0[c], s0);
      s1 = wmma_bf16(aq[c], bk1[c], s1);
    }

    // ---- causal mask + online softmax ----
    float pr0[8], pr1[8];
    #pragma unroll
    for (int r = 0; r < 8; ++r) {
      const int qrowg = mTile + r + 8 * half;
      float v0 = (kk + l16      <= qrowg) ? s0[r] : -1.0e9f;
      float v1 = (kk + 16 + l16 <= qrowg) ? s1[r] : -1.0e9f;
      float mx = fmaxf(v0, v1);
      #pragma unroll
      for (int off = 8; off > 0; off >>= 1) mx = fmaxf(mx, __shfl_xor(mx, off, 32));
      const float mnew = fmaxf(mrow[r], mx);
      const float corr = __expf(mrow[r] - mnew);
      const float p0 = __expf(v0 - mnew);
      const float p1 = __expf(v1 - mnew);
      float sum = p0 + p1;
      #pragma unroll
      for (int off = 8; off > 0; off >>= 1) sum += __shfl_xor(sum, off, 32);
      mrow[r] = mnew;
      lrow[r] = lrow[r] * corr + sum;
      #pragma unroll
      for (int t = 0; t < 8; ++t) oacc[t][r] *= corr;
      pr0[r] = p0; pr1[r] = p1;
    }

    // ---- P: C/D layout -> A layout via LDS ----
    __syncthreads();
    #pragma unroll
    for (int r = 0; r < 8; ++r) {
      lp[wave][r + 8 * half][l16]      = (__bf16)pr0[r];
      lp[wave][r + 8 * half][16 + l16] = (__bf16)pr1[r];
    }
    __syncthreads();
    v16bf ap;
    #pragma unroll
    for (int j = 0; j < 8; ++j) {
      ap[j]     = lp[wave][l16][8 * half + j];
      ap[j + 8] = lp[wave][l16][16 + 8 * half + j];
    }

    // ---- O += P (16x32) x V_chunk (32x128) ----
    #pragma unroll
    for (int t = 0; t < 8; ++t) {
      const __bf16* vr = &Vs[buf][t * 16 + l16][8 * half];
      v16bf bv;
      #pragma unroll
      for (int j = 0; j < 8; ++j) { bv[j] = vr[j]; bv[j + 8] = vr[16 + j]; }
      oacc[t] = wmma_bf16(ap, bv, oacc[t]);
    }

    publish();   // next K/V slab landed; everyone done with current slab
    buf ^= 1;
  }

  // ---- normalize and store bf16 in [s][h*128 + d] layout ----
  #pragma unroll
  for (int r = 0; r < 8; ++r) {
    const float inv = 1.f / lrow[r];
    const int srow = mTile + r + 8 * half;
    #pragma unroll
    for (int t = 0; t < 8; ++t)
      Ab[(size_t)srow * MODEL + h * HDIM + t * 16 + l16] =
          (__bf16)(oacc[t][r] * inv);
  }
}

// -----------------------------------------------------------------------------
extern "C" void kernel_launch(void* const* d_in, const int* in_sizes, int n_in,
                              void* d_out, int out_size, void* d_ws, size_t ws_size,
                              hipStream_t stream)
{
  const float* X    = (const float*)d_in[0];
  // d_in[1] = attention_mask: pure causal, applied analytically in-kernel
  const float* cosb = (const float*)d_in[2];
  const float* sinb = (const float*)d_in[3];
  const float* Wq   = (const float*)d_in[4];
  const float* Wk   = (const float*)d_in[5];
  const float* Wv   = (const float*)d_in[6];
  const float* Wo   = (const float*)d_in[7];
  float* out = (float*)d_out;

  char* ws = (char*)d_ws;
  size_t off = 0;
  float*  Qf = (float*)(ws + off);  off += (size_t)S_LEN * MODEL   * sizeof(float);
  float*  Kf = (float*)(ws + off);  off += (size_t)S_LEN * KVMODEL * sizeof(float);
  float*  Vf = (float*)(ws + off);  off += (size_t)S_LEN * KVMODEL * sizeof(float);
  __bf16* Qb = (__bf16*)(ws + off); off += (size_t)S_LEN * MODEL   * sizeof(__bf16);
  __bf16* Kb = (__bf16*)(ws + off); off += (size_t)S_LEN * KVMODEL * sizeof(__bf16);
  __bf16* Vt = (__bf16*)(ws + off); off += (size_t)S_LEN * KVMODEL * sizeof(__bf16);
  __bf16* Ab = (__bf16*)(ws + off); off += (size_t)S_LEN * MODEL   * sizeof(__bf16);
  (void)ws_size; (void)in_sizes; (void)n_in; (void)out_size;

  const dim3 blk(256);

  // Q/K/V projections: out[s,o] = sum_i X[s,i] * W[o,i]
  gemm_bt_tiled_kernel<float><<<dim3(MODEL   / 128, S_LEN / 128), blk, 0, stream>>>(X, Wq, Qf, S_LEN, MODEL,   MODEL);
  gemm_bt_tiled_kernel<float><<<dim3(KVMODEL / 128, S_LEN / 128), blk, 0, stream>>>(X, Wk, Kf, S_LEN, KVMODEL, MODEL);
  gemm_bt_tiled_kernel<float><<<dim3(KVMODEL / 128, S_LEN / 128), blk, 0, stream>>>(X, Wv, Vf, S_LEN, KVMODEL, MODEL);

  {
    const int total = S_LEN * (NHEADS + KVHEADS) * (HDIM / 2);
    rope_convert_kernel<<<(total + 255) / 256, blk, 0, stream>>>(
        Qf, Kf, cosb, sinb, Qb, Kb, 0.08838834764831845f /* 1/sqrt(128) */);
  }
  {
    const int total = S_LEN * KVMODEL;
    v_transpose_kernel<<<(total + 255) / 256, blk, 0, stream>>>(Vf, Vt);
  }

  flash_attn_kernel<<<dim3(S_LEN / 64, NHEADS), dim3(128), 0, stream>>>(Qb, Kb, Vt, Ab);

  // out[s,p] = sum_o attn[s,o] * Wo[p,o]
  gemm_bt_tiled_kernel<__bf16><<<dim3(MODEL / 128, S_LEN / 128), blk, 0, stream>>>(Ab, Wo, out, S_LEN, MODEL, MODEL);
}
